// MachineBranch_35562329211643
// MI455X (gfx1250) — matrix operand order repaired and checked by reading, not compiled, not run on hardware
//
#include <hip/hip_runtime.h>
#include <math.h>

// Problem constants (match reference)
#define BB   4
#define MM   64
#define NN   512
#define DD   256
#define HH   8
#define DKK  32
#define FF   1024
#define SCALE_F 0.17677669529663687f   // 1/sqrt(32)

typedef __attribute__((ext_vector_type(16))) __bf16          bf16x16;
typedef __attribute__((ext_vector_type(16))) unsigned short  u16x16;
typedef __attribute__((ext_vector_type(8)))  float           f32x8;

__device__ __forceinline__ unsigned short f32_to_bf16(float f) {
  unsigned int x = __builtin_bit_cast(unsigned int, f);
  x += 0x7fffu + ((x >> 16) & 1u);           // round-to-nearest-even
  return (unsigned short)(x >> 16);
}

__device__ __forceinline__ unsigned int pack_bf16x2(float a, float b) {
  return (unsigned int)f32_to_bf16(a) | ((unsigned int)f32_to_bf16(b) << 16);
}

__device__ __forceinline__ u16x16 make_u16x16(uint4 lo, uint4 hi) {
  struct Pair { uint4 lo, hi; } p{lo, hi};
  return __builtin_bit_cast(u16x16, p);
}

__device__ __forceinline__ f32x8 wmma_bf16(u16x16 a, u16x16 b, f32x8 c) {
  return __builtin_amdgcn_wmma_f32_16x16x32_bf16(
      false, __builtin_bit_cast(bf16x16, a),
      false, __builtin_bit_cast(bf16x16, b),
      (short)0, c, false, false);
}

// ---------------------------------------------------------------------------
// fp32 W[K x C] -> bf16 W^T[C x K]  (coalesced writes)
// ---------------------------------------------------------------------------
__global__ void cvt_bf16_T_kernel(const float* __restrict__ src,
                                  unsigned short* __restrict__ dst,
                                  int K, int C) {
  int i = blockIdx.x * blockDim.x + threadIdx.x;   // index into dst (C*K)
  if (i < K * C) {
    int c = i / K, k = i - c * K;
    dst[i] = f32_to_bf16(src[(size_t)k * C + c]);
  }
}

// ---------------------------------------------------------------------------
// Row LayerNorm (D=256), one 256-thread block per row, bf16 output
// ---------------------------------------------------------------------------
__global__ __launch_bounds__(256) void ln_kernel(
    const float* __restrict__ x, const float* __restrict__ g,
    const float* __restrict__ bta, unsigned short* __restrict__ out_bf) {
  __shared__ float red[256];
  const int row = blockIdx.x, t = threadIdx.x;
  float v = x[(size_t)row * DD + t];
  red[t] = v; __syncthreads();
  for (int s = 128; s > 0; s >>= 1) { if (t < s) red[t] += red[t + s]; __syncthreads(); }
  float mu = red[0] * (1.0f / DD);
  __syncthreads();
  float dv = v - mu;
  red[t] = dv * dv; __syncthreads();
  for (int s = 128; s > 0; s >>= 1) { if (t < s) red[t] += red[t + s]; __syncthreads(); }
  float var = red[0] * (1.0f / DD);
  float y = dv * rsqrtf(var + 1e-5f) * g[t] + bta[t];
  out_bf[(size_t)row * DD + t] = f32_to_bf16(y);
}

// residual add + LayerNorm: sum = a+b (kept fp32), ln(sum) -> bf16
__global__ __launch_bounds__(256) void add_ln_kernel(
    const float* __restrict__ a, const float* __restrict__ bsrc,
    const float* __restrict__ g, const float* __restrict__ bta,
    float* __restrict__ sum_out, unsigned short* __restrict__ ln_bf) {
  __shared__ float red[256];
  const int row = blockIdx.x, t = threadIdx.x;
  const size_t idx = (size_t)row * DD + t;
  float v = a[idx] + bsrc[idx];
  sum_out[idx] = v;
  red[t] = v; __syncthreads();
  for (int s = 128; s > 0; s >>= 1) { if (t < s) red[t] += red[t + s]; __syncthreads(); }
  float mu = red[0] * (1.0f / DD);
  __syncthreads();
  float dv = v - mu;
  red[t] = dv * dv; __syncthreads();
  for (int s = 128; s > 0; s >>= 1) { if (t < s) red[t] += red[t + s]; __syncthreads(); }
  float var = red[0] * (1.0f / DD);
  float y = dv * rsqrtf(var + 1e-5f) * g[t] + bta[t];
  ln_bf[idx] = f32_to_bf16(y);
}

__global__ void add_kernel(const float* __restrict__ a, const float* __restrict__ b,
                           float* __restrict__ o, int n) {
  int i = blockIdx.x * blockDim.x + threadIdx.x;
  if (i < n) o[i] = a[i] + b[i];
}

// ---------------------------------------------------------------------------
// Generic bf16 GEMM with TRANSPOSED weights:
//   C[R x Ccols] = A[R x K] @ W[K x Ccols] + bias,  W given as Wt[Ccols x K].
// One wave per 16x16 tile; A and B fragments are each two b128 loads per lane.
// ---------------------------------------------------------------------------
__global__ __launch_bounds__(32) void gemm_bf16_wmma_kernel(
    const unsigned short* __restrict__ A, const unsigned short* __restrict__ Wt,
    const float* __restrict__ bias, float* __restrict__ Cf32,
    unsigned short* __restrict__ Cbf16, int R, int K, int Ccols, int relu) {
  const int lane = threadIdx.x;
  const int hl = lane >> 4, l16 = lane & 15;
  const int ct = blockIdx.x * 16;
  const int rt = blockIdx.y * 16;
  const unsigned short* arow = A + (size_t)(rt + l16) * K + hl * 8;
  const unsigned short* brow = Wt + (size_t)(ct + l16) * K + hl * 8;
  f32x8 acc = {};
  for (int k0 = 0; k0 < K; k0 += 32) {
    uint4 a0 = *(const uint4*)(arow + k0);
    uint4 a1 = *(const uint4*)(arow + k0 + 16);
    uint4 b0 = *(const uint4*)(brow + k0);
    uint4 b1 = *(const uint4*)(brow + k0 + 16);
    acc = wmma_bf16(make_u16x16(a0, a1), make_u16x16(b0, b1), acc);
  }
  const int col = ct + l16;
  const float bv = bias ? bias[col] : 0.f;
#pragma unroll
  for (int v = 0; v < 8; ++v) {
    const int row = rt + v + 8 * hl;
    float val = acc[v] + bv;
    if (relu) val = fmaxf(val, 0.f);
    if (Cf32)  Cf32[(size_t)row * Ccols + col] = val;
    if (Cbf16) Cbf16[(size_t)row * Ccols + col] = f32_to_bf16(val);
  }
}

// ---------------------------------------------------------------------------
// Fused edge-projection + flash attention. One 256-thread WG per (b,m).
// LDS: We^T (bf16, padded rows) resident; per 16-row n-tile: edge bf16 tile,
// projected e (fp32), scores, online-softmax state.
// Edge tiles are register double-buffered: the next tile's global loads are
// issued right after staging the current tile so HBM latency hides under the
// WMMA projection + softmax work.
// ---------------------------------------------------------------------------
#define KP  (DD + 8)   // padded LDS row stride (halves) for We^T / edge tile
#define EPF (DD + 8)   // padded LDS row stride (floats) for e

__global__ __launch_bounds__(256) void attn_fused_kernel(
    const float* __restrict__ edge,             // (B,N,M,D) fp32
    const unsigned short* __restrict__ WeT_bf,  // (D,D) bf16, transposed (out x in)
    const float* __restrict__ be,
    const float* __restrict__ qbuf,             // (B*M, D) fp32
    const float* __restrict__ ksbuf,            // (B*M, D)
    const float* __restrict__ vsbuf,            // (B*M, D)
    const float* __restrict__ kobuf,            // (B*N, D)
    const float* __restrict__ vobuf,            // (B*N, D)
    const unsigned char* __restrict__ mask,     // (B,N,M) bool (1 byte)
    unsigned short* __restrict__ attn_bf)       // (B*M, D) bf16
{
  extern __shared__ char smem[];
  unsigned short* sWeT  = (unsigned short*)smem;        // DD rows x KP halves
  unsigned short* sEdge = sWeT + DD * KP;               // 16 rows x KP halves
  float* sE     = (float*)(sEdge + 16 * KP);            // 16 rows x EPF floats
  float* sQ     = sE + 16 * EPF;                        // 256
  float* sScore = sQ + DD;                              // 8*16
  float* sMx    = sScore + HH * 16;                     // 8
  float* sSm    = sMx + HH;                             // 8
  float* sAcc   = sSm + HH;                             // 8*32

  const int tid = threadIdx.x;
  const int bm  = blockIdx.x;
  const int b   = bm / MM;
  const int m   = bm % MM;

  // --- edge tile register double-buffer setup (16 floats per thread) ---
  const int er = tid >> 4, ec0 = (tid & 15) << 4;
  const float* ebase = edge + ((size_t)((b * NN + er) * MM + m)) * DD + ec0;
  const size_t estep = (size_t)MM * DD;   // advance by one n
  float4 f0, f1, f2, f3;
  { // prefetch tile n0 = 0
    const float* s = ebase;
    f0 = *(const float4*)(s + 0);  f1 = *(const float4*)(s + 4);
    f2 = *(const float4*)(s + 8);  f3 = *(const float4*)(s + 12);
  }

  { // We^T (bf16) -> LDS with padded rows, 16B chunks
    const uint4* src = (const uint4*)WeT_bf;
    for (int i = tid; i < DD * (DD / 8); i += 256) {
      const int row = i >> 5, c8 = i & 31;   // DD/8 = 32 chunks per row
      *(uint4*)(sWeT + row * KP + c8 * 8) = src[i];
    }
  }
  sQ[tid] = qbuf[(size_t)bm * DD + tid];
  if (tid < HH) { sMx[tid] = -INFINITY; sSm[tid] = 0.f; }
  sAcc[tid] = 0.f;
  __syncthreads();

  const int wave = tid >> 5, lane = tid & 31;
  const int hl = lane >> 4, l16 = lane & 15;

  for (int n0 = 0; n0 < NN; n0 += 16) {
    { // stage current tile (registers) -> bf16 LDS tile, 2x b128 stores
      unsigned short* dst = sEdge + er * KP + ec0;
      uint4 o0 = {pack_bf16x2(f0.x, f0.y), pack_bf16x2(f0.z, f0.w),
                  pack_bf16x2(f1.x, f1.y), pack_bf16x2(f1.z, f1.w)};
      uint4 o1 = {pack_bf16x2(f2.x, f2.y), pack_bf16x2(f2.z, f2.w),
                  pack_bf16x2(f3.x, f3.y), pack_bf16x2(f3.z, f3.w)};
      *(uint4*)(dst)     = o0;
      *(uint4*)(dst + 8) = o1;
    }
    __syncthreads();

    if (n0 + 16 < NN) { // issue next tile's loads; latency hides under compute
      const float* s = ebase + (size_t)(n0 + 16) * estep;
      f0 = *(const float4*)(s + 0);  f1 = *(const float4*)(s + 4);
      f2 = *(const float4*)(s + 8);  f3 = *(const float4*)(s + 12);
    }

    // e[16 x 256] = edge_tile @ We + be  (wave w covers cols [32w, 32w+32))
#pragma unroll
    for (int sub = 0; sub < 2; ++sub) {
      const int ct = wave * 32 + sub * 16;
      const unsigned short* arow = sEdge + l16 * KP + hl * 8;
      const unsigned short* brow = sWeT + (ct + l16) * KP + hl * 8;
      f32x8 acc = {};
#pragma unroll
      for (int k0 = 0; k0 < DD; k0 += 32) {
        uint4 a0 = *(const uint4*)(arow + k0);
        uint4 a1 = *(const uint4*)(arow + k0 + 16);
        uint4 b0 = *(const uint4*)(brow + k0);
        uint4 b1 = *(const uint4*)(brow + k0 + 16);
        acc = wmma_bf16(make_u16x16(a0, a1), make_u16x16(b0, b1), acc);
      }
      const float bias = be[ct + l16];
#pragma unroll
      for (int v = 0; v < 8; ++v)
        sE[(v + 8 * hl) * EPF + ct + l16] = acc[v] + bias;
    }
    __syncthreads();

    // scores[h][i] = SCALE * sum_dk (q+e)(k+e), masked by o2m_mask[b,n,m]
    if (tid < HH * 16) {
      const int h = tid >> 4, i = tid & 15;
      const float* e = sE + i * EPF + h * DKK;
      const float* k = kobuf + ((size_t)(b * NN + n0 + i)) * DD + h * DKK;
      const float* q = sQ + h * DKK;
      float s = 0.f;
#pragma unroll
      for (int d = 0; d < DKK; ++d) { float ed = e[d]; s += (q[d] + ed) * (k[d] + ed); }
      s *= SCALE_F;
      sScore[tid] = mask[(size_t)(b * NN + n0 + i) * MM + m] ? s : -INFINITY;
    }
    __syncthreads();

    // online softmax update: one wave per head, thread d owns dim d
    {
      const int h = tid >> 5, d = tid & 31;
      const float* sc = sScore + h * 16;
      float tmax = -INFINITY;
#pragma unroll
      for (int i = 0; i < 16; ++i) tmax = fmaxf(tmax, sc[i]);
      const float mx = sMx[h];
      const float nmx = fmaxf(mx, tmax);
      if (nmx != -INFINITY) {
        const float sf = __expf(mx - nmx);      // exp(-inf)=0 when mx==-inf
        float psum = 0.f, accd = sAcc[h * DKK + d] * sf;
#pragma unroll
        for (int i = 0; i < 16; ++i) {
          float p = __expf(sc[i] - nmx);
          psum += p;
          float vv = vobuf[(size_t)(b * NN + n0 + i) * DD + h * DKK + d]
                   + sE[i * EPF + h * DKK + d];
          accd = fmaf(p, vv, accd);
        }
        sAcc[h * DKK + d] = accd;
        if (d == 0) { sMx[h] = nmx; sSm[h] = sSm[h] * sf + psum; }
      }
    }
    __syncthreads();
  }

  // self token (always unmasked) + finalize attn_out
  {
    const int h = tid >> 5, d = tid & 31;
    const float* q = sQ + h * DKK;
    const float* ks = ksbuf + (size_t)bm * DD + h * DKK;
    float s = 0.f;
#pragma unroll
    for (int dd = 0; dd < DKK; ++dd) s = fmaf(q[dd], ks[dd], s);
    s *= SCALE_F;
    const float mx = sMx[h];
    const float nmx = fmaxf(mx, s);
    const float sf = (mx == -INFINITY) ? 0.f : __expf(mx - nmx);
    const float p = __expf(s - nmx);
    const float accd = sAcc[h * DKK + d] * sf + p * vsbuf[(size_t)bm * DD + h * DKK + d];
    const float sm = sSm[h] * sf + p;
    attn_bf[(size_t)bm * DD + h * DKK + d] = f32_to_bf16(accd / sm);
  }
}

// ---------------------------------------------------------------------------
extern "C" void kernel_launch(void* const* d_in, const int* in_sizes, int n_in,
                              void* d_out, int out_size, void* d_ws, size_t ws_size,
                              hipStream_t stream) {
  (void)in_sizes; (void)n_in; (void)out_size; (void)ws_size;
  const float* machine = (const float*)d_in[0];
  const float* opemb   = (const float*)d_in[1];
  const float* edge    = (const float*)d_in[2];
  const unsigned char* mask = (const unsigned char*)d_in[3];  // jax bool -> 1 byte
  const float* Wq = (const float*)d_in[4],  *bq = (const float*)d_in[5];
  const float* Wk = (const float*)d_in[6],  *bk = (const float*)d_in[7];
  const float* Wv = (const float*)d_in[8],  *bv = (const float*)d_in[9];
  const float* Wo = (const float*)d_in[10], *bo = (const float*)d_in[11];
  const float* Wks = (const float*)d_in[12], *bks = (const float*)d_in[13];
  const float* Wvs = (const float*)d_in[14], *bvs = (const float*)d_in[15];
  const float* We = (const float*)d_in[16], *be = (const float*)d_in[17];
  const float* g1 = (const float*)d_in[18], *bn1 = (const float*)d_in[19];
  const float* g2 = (const float*)d_in[20], *bn2 = (const float*)d_in[21];
  const float* gop = (const float*)d_in[22], *bnop = (const float*)d_in[23];
  const float* Wf1 = (const float*)d_in[24], *bf1 = (const float*)d_in[25];
  const float* Wf2 = (const float*)d_in[26], *bf2 = (const float*)d_in[27];
  float* out = (float*)d_out;

  char* ws = (char*)d_ws;
  size_t off = 0;
  auto alloc = [&](size_t bytes) {
    char* p = ws + off; off += (bytes + 255) & ~(size_t)255; return p;
  };
  unsigned short* mnorm_bf = (unsigned short*)alloc((size_t)BB * MM * DD * 2);
  unsigned short* onorm_bf = (unsigned short*)alloc((size_t)BB * NN * DD * 2);
  unsigned short* WqT  = (unsigned short*)alloc((size_t)DD * DD * 2);
  unsigned short* WkT  = (unsigned short*)alloc((size_t)DD * DD * 2);
  unsigned short* WvT  = (unsigned short*)alloc((size_t)DD * DD * 2);
  unsigned short* WoT  = (unsigned short*)alloc((size_t)DD * DD * 2);
  unsigned short* WksT = (unsigned short*)alloc((size_t)DD * DD * 2);
  unsigned short* WvsT = (unsigned short*)alloc((size_t)DD * DD * 2);
  unsigned short* WeT  = (unsigned short*)alloc((size_t)DD * DD * 2);
  unsigned short* Wf1T = (unsigned short*)alloc((size_t)DD * FF * 2);
  unsigned short* Wf2T = (unsigned short*)alloc((size_t)FF * DD * 2);
  float* qbuf  = (float*)alloc((size_t)BB * MM * DD * 4);
  float* ksbuf = (float*)alloc((size_t)BB * MM * DD * 4);
  float* vsbuf = (float*)alloc((size_t)BB * MM * DD * 4);
  float* kobuf = (float*)alloc((size_t)BB * NN * DD * 4);
  float* vobuf = (float*)alloc((size_t)BB * NN * DD * 4);
  unsigned short* attn_bf = (unsigned short*)alloc((size_t)BB * MM * DD * 2);
  float* attnproj = (float*)alloc((size_t)BB * MM * DD * 4);
  float* mout     = (float*)alloc((size_t)BB * MM * DD * 4);
  unsigned short* h_bf    = (unsigned short*)alloc((size_t)BB * MM * DD * 2);
  unsigned short* ffn1_bf = (unsigned short*)alloc((size_t)BB * MM * FF * 2);
  float* ffn2     = (float*)alloc((size_t)BB * MM * DD * 4);

  // 1) LayerNorms -> bf16 activations
  ln_kernel<<<BB * MM, 256, 0, stream>>>(machine, g1, bn1, mnorm_bf);
  ln_kernel<<<BB * NN, 256, 0, stream>>>(opemb, gop, bnop, onorm_bf);

  // 2) weights -> bf16, transposed (out x in)
  auto cvtT = [&](const float* s, unsigned short* d, int K, int C) {
    int n = K * C;
    cvt_bf16_T_kernel<<<(n + 255) / 256, 256, 0, stream>>>(s, d, K, C);
  };
  cvtT(Wq, WqT, DD, DD);   cvtT(Wk, WkT, DD, DD);
  cvtT(Wv, WvT, DD, DD);   cvtT(Wo, WoT, DD, DD);
  cvtT(Wks, WksT, DD, DD); cvtT(Wvs, WvsT, DD, DD);
  cvtT(We, WeT, DD, DD);   cvtT(Wf1, Wf1T, DD, FF);
  cvtT(Wf2, Wf2T, FF, DD);

  // 3) projections (WMMA GEMMs)
  gemm_bf16_wmma_kernel<<<dim3(DD / 16, BB * MM / 16), 32, 0, stream>>>(
      mnorm_bf, WqT, bq, qbuf, nullptr, BB * MM, DD, DD, 0);
  gemm_bf16_wmma_kernel<<<dim3(DD / 16, BB * MM / 16), 32, 0, stream>>>(
      mnorm_bf, WksT, bks, ksbuf, nullptr, BB * MM, DD, DD, 0);
  gemm_bf16_wmma_kernel<<<dim3(DD / 16, BB * MM / 16), 32, 0, stream>>>(
      mnorm_bf, WvsT, bvs, vsbuf, nullptr, BB * MM, DD, DD, 0);
  gemm_bf16_wmma_kernel<<<dim3(DD / 16, BB * NN / 16), 32, 0, stream>>>(
      onorm_bf, WkT, bk, kobuf, nullptr, BB * NN, DD, DD, 0);
  gemm_bf16_wmma_kernel<<<dim3(DD / 16, BB * NN / 16), 32, 0, stream>>>(
      onorm_bf, WvT, bv, vobuf, nullptr, BB * NN, DD, DD, 0);

  // 4) fused edge projection + flash attention
  constexpr size_t SMEM = (size_t)(DD * KP + 16 * KP) * 2 +
                          (size_t)(16 * EPF + DD + HH * 16 + HH + HH + HH * DKK) * 4;
  attn_fused_kernel<<<BB * MM, 256, SMEM, stream>>>(
      edge, WeT, be, qbuf, ksbuf, vsbuf, kobuf, vobuf, mask, attn_bf);

  // 5) output projection, residual + LN2, FFN, final residual
  gemm_bf16_wmma_kernel<<<dim3(DD / 16, BB * MM / 16), 32, 0, stream>>>(
      attn_bf, WoT, bo, attnproj, nullptr, BB * MM, DD, DD, 0);
  add_ln_kernel<<<BB * MM, 256, 0, stream>>>(machine, attnproj, g2, bn2, mout, h_bf);
  gemm_bf16_wmma_kernel<<<dim3(FF / 16, BB * MM / 16), 32, 0, stream>>>(
      h_bf, Wf1T, bf1, nullptr, ffn1_bf, BB * MM, DD, FF, 1);
  gemm_bf16_wmma_kernel<<<dim3(DD / 16, BB * MM / 16), 32, 0, stream>>>(
      ffn1_bf, Wf2T, bf2, ffn2, nullptr, BB * MM, FF, DD, 0);
  add_kernel<<<BB * MM * DD / 256, 256, 0, stream>>>(mout, ffn2, out, BB * MM * DD);
}
